// LSTM_49770081026750
// MI455X (gfx1250) — matrix operand order repaired and checked by reading, not compiled, not run on hardware
//
#include <hip/hip_runtime.h>
#include <stdint.h>

// ---------------------------------------------------------------------------
// 2-layer LSTM, B=32, T=512, I=H=1024, bf16 WMMA with fp32 accumulate.
//
// One fused kernel per (layer, timestep):
//   - 64 h-tiles x 2 batch-tiles = 128 tile-groups; each tile-group owns all
//     FOUR gates (f,w,i,o) of a 16(h) x 16(b) tile -> 4 independent WMMA
//     accumulator chains sharing the B fragments.
//   - split-K x4 (each wave covers K=256) => 512 waves / 64 blocks keep enough
//     loads in flight to stream the 16.8MB L2-resident weight slice.
//   - partials reduced through LDS (ds_store_b128/ds_load_b128 + s_barrier);
//     the ks==0 wave adds bias and performs the LSTM cell update in-register,
//     storing c, h(fp32) and h(bf16) with 128-bit stores.
// ---------------------------------------------------------------------------

typedef __attribute__((ext_vector_type(16))) __bf16 v16bf;
typedef __attribute__((ext_vector_type(8)))  float  v8f;

union ABFrag { v16bf v; uint4 q[2]; };
union AccU   { v8f v; float4 f4[2]; float f[8]; };

static __device__ __forceinline__ uint16_t f2bf(float f) {
  uint32_t u = __builtin_bit_cast(uint32_t, f);
  u += 0x7FFFu + ((u >> 16) & 1u);           // round-to-nearest-even
  return (uint16_t)(u >> 16);
}

static __device__ __forceinline__ uint32_t pack2bf(float a, float b) {
  uint32_t ua = __builtin_bit_cast(uint32_t, a);
  ua += 0x7FFFu + ((ua >> 16) & 1u);
  uint32_t ub = __builtin_bit_cast(uint32_t, b);
  ub += 0x7FFFu + ((ub >> 16) & 1u);
  return (ua >> 16) | (ub & 0xFFFF0000u);
}

static __device__ __forceinline__ float sigf(float x) {
  return 1.f / (1.f + __expf(-x));
}

// ---- prep kernels ---------------------------------------------------------
__global__ __launch_bounds__(256) void k_f32_to_bf16(const float* __restrict__ in,
                                                     uint16_t* __restrict__ out,
                                                     int n) {
  int i = blockIdx.x * blockDim.x + threadIdx.x;
  if (i < n) out[i] = f2bf(in[i]);
}

__global__ __launch_bounds__(256) void k_zero_u32(uint32_t* __restrict__ p, int n) {
  int i = blockIdx.x * blockDim.x + threadIdx.x;
  if (i < n) p[i] = 0u;
}

// ---- fused per-(layer,step) kernel: gates GEMM + LSTM cell ----------------
__global__ __launch_bounds__(256) void k_lstm_layer_step(
    const uint16_t* __restrict__ Wh,     // [4][1024][1024] bf16 (layer slice)
    const uint16_t* __restrict__ Wx,     // [4][1024][1024] bf16 (layer slice)
    const uint16_t* __restrict__ hprev,  // [32][1024] bf16 (read buffer)
    const uint16_t* __restrict__ xin,    // rows of 1024, stride xstride
    long xstride,
    const float* __restrict__ bias,      // [4][1024]
    float* __restrict__ c,               // [32][1024] state (in/out)
    float* __restrict__ hout,            // [32][1024] fp32 (out)
    uint16_t* __restrict__ hbfout)       // [32][1024] bf16 (write buffer)
{
  constexpr int H = 1024;
  __shared__ float smem[8 * 4 * 32 * 8];   // [wave][gate][lane][r], 32 KB

  const int lane  = threadIdx.x & 31;
  const int ww    = threadIdx.x >> 5;      // wave in block: 0..7
  const int tgl   = ww >> 2;               // tile-group in block: 0..1
  const int ks    = ww & 3;                // K-split id: 0..3
  const int tg    = blockIdx.x * 2 + tgl;  // 0..127
  const int htile = tg >> 1;               // 0..63 (16 h-rows each)
  const int ntile = tg & 1;                // batch halves
  const int half  = lane >> 4;
  const int l16   = lane & 15;

  const int hrow0 = htile * 16 + l16;      // A row within one gate's block
  const int bcol  = ntile * 16 + l16;      // batch column (B/C/D N index)

  const uint16_t* __restrict__ whrow = Wh + (size_t)hrow0 * H;
  const uint16_t* __restrict__ wxrow = Wx + (size_t)hrow0 * H;
  const uint16_t* __restrict__ hrow  = hprev + (size_t)bcol * H;
  const uint16_t* __restrict__ xrow  = xin + (size_t)bcol * xstride;

  AccU acc[4];
#pragma unroll
  for (int g = 0; g < 4; ++g) acc[g].v = v8f{};

  const int k0 = ks * 256;                 // this wave's K chunk (8 k-tiles)
#pragma unroll 2
  for (int kt = 0; kt < 8; ++kt) {
    const int kk  = k0 + kt * 32;
    const int kA0 = kk + half * 8;         // A frag: v0..3=K..K+7, v4..7=K+16..23
    const int kB0 = kk + half * 8;         // B frag: v0..3=K..K+7, v4..7=K+16..23

    __builtin_prefetch(whrow + kA0 + 32, 0, 1);
    __builtin_prefetch(wxrow + kA0 + 32, 0, 1);

    ABFrag bh, bx;
    bh.q[0] = *(const uint4*)(hrow + kB0);
    bh.q[1] = *(const uint4*)(hrow + kB0 + 16);
    bx.q[0] = *(const uint4*)(xrow + kB0);
    bx.q[1] = *(const uint4*)(xrow + kB0 + 16);

#pragma unroll
    for (int g = 0; g < 4; ++g) {          // gate blocks at +g*2MB (imm offsets)
      const size_t go = (size_t)g * ((size_t)H * H);
      ABFrag a;
      a.q[0] = *(const uint4*)(whrow + go + kA0);
      a.q[1] = *(const uint4*)(whrow + go + kA0 + 16);
      acc[g].v = __builtin_amdgcn_wmma_f32_16x16x32_bf16(
          false, a.v, false, bh.v, (short)0, acc[g].v, false, false);
      a.q[0] = *(const uint4*)(wxrow + go + kA0);
      a.q[1] = *(const uint4*)(wxrow + go + kA0 + 16);
      acc[g].v = __builtin_amdgcn_wmma_f32_16x16x32_bf16(
          false, a.v, false, bx.v, (short)0, acc[g].v, false, false);
    }
  }

  // ---- split-K reduction through LDS (ks!=0 publish, ks==0 consume) ----
  if (ks != 0) {
#pragma unroll
    for (int g = 0; g < 4; ++g) {
      float* p = &smem[(size_t)(((ww * 4) + g) * 32 + lane) * 8];
      *(float4*)(p)     = acc[g].f4[0];
      *(float4*)(p + 4) = acc[g].f4[1];
    }
  }
  __syncthreads();
  if (ks != 0) return;

#pragma unroll
  for (int g = 0; g < 4; ++g) {
#pragma unroll
    for (int j = 1; j < 4; ++j) {
      const float* p =
          &smem[(size_t)((((tgl * 4 + j) * 4) + g) * 32 + lane) * 8];
      AccU t;
      t.f4[0] = *(const float4*)(p);
      t.f4[1] = *(const float4*)(p + 4);
      acc[g].v = acc[g].v + t.v;
    }
  }

  // ---- bias + fused LSTM cell (C/D layout: M = hb + r, N = bcol) ----
  const int hb = htile * 16 + half * 8;    // first of this lane's 8 h rows
#pragma unroll
  for (int g = 0; g < 4; ++g) {
    const float* bp = bias + g * H + hb;
    AccU bb;
    bb.f4[0] = *(const float4*)(bp);
    bb.f4[1] = *(const float4*)(bp + 4);
    acc[g].v = acc[g].v + bb.v;
  }

  float* cp = c + (size_t)bcol * H + hb;
  AccU cold, cnew, hnew;
  cold.f4[0] = *(const float4*)(cp);
  cold.f4[1] = *(const float4*)(cp + 4);

#pragma unroll
  for (int r = 0; r < 8; ++r) {
    const float f = acc[0].f[r];
    const float w = acc[1].f[r];
    const float i = acc[2].f[r];
    const float o = acc[3].f[r];
    const float cn = cold.f[r] * sigf(f) + sigf(w) * tanhf(i);
    const float hn = tanhf(cn) * sigf(o);
    cnew.f[r] = cn;
    hnew.f[r] = hn;
  }

  *(float4*)(cp)     = cnew.f4[0];
  *(float4*)(cp + 4) = cnew.f4[1];

  float* hp = hout + (size_t)bcol * H + hb;
  *(float4*)(hp)     = hnew.f4[0];
  *(float4*)(hp + 4) = hnew.f4[1];

  uint4 pk;
  pk.x = pack2bf(hnew.f[0], hnew.f[1]);
  pk.y = pack2bf(hnew.f[2], hnew.f[3]);
  pk.z = pack2bf(hnew.f[4], hnew.f[5]);
  pk.w = pack2bf(hnew.f[6], hnew.f[7]);
  *(uint4*)(hbfout + (size_t)bcol * H + hb) = pk;
}

// ---------------------------------------------------------------------------
extern "C" void kernel_launch(void* const* d_in, const int* in_sizes, int n_in,
                              void* d_out, int out_size, void* d_ws, size_t ws_size,
                              hipStream_t stream) {
  (void)in_sizes; (void)n_in; (void)out_size; (void)ws_size;
  constexpr int Bn = 32, T = 512, I = 1024, H = 1024;
  constexpr int BH = Bn * H;                 // 32768 state elements per layer

  const float* seq  = (const float*)d_in[0];  // (32, 512, 1024)
  const float* Wh   = (const float*)d_in[1];  // (2, 4, 1024, 1024)
  const float* Wx   = (const float*)d_in[2];  // (2, 4, 1024, 1024)
  const float* bias = (const float*)d_in[3];  // (2, 4, 1024)

  // ---- workspace layout (~64.8 MB, all regions 256B aligned) ----
  const size_t nW   = (size_t)2 * 4 * H * H;   // 8,388,608 per weight tensor
  const size_t nSeq = (size_t)Bn * T * I;      // 16,777,216
  char* ws = (char*)d_ws;
  uint16_t* whbf  = (uint16_t*)ws;  ws += nW * 2;               // 16 MB
  uint16_t* wxbf  = (uint16_t*)ws;  ws += nW * 2;               // 16 MB
  uint16_t* seqbf = (uint16_t*)ws;  ws += nSeq * 2;             // 32 MB
  float*    cst   = (float*)ws;     ws += (size_t)2 * BH * 4;   // c, 2 layers
  uint16_t* hbfb  = (uint16_t*)ws;  ws += (size_t)4 * BH * 2;   // h bf16, 2 layers x ping-pong
  float*    hst   = (float*)ws;                                 // h fp32, 2 layers

  // ---- one-time converts (weights become L2-resident bf16) ----
  k_f32_to_bf16<<<(int)(nW / 256), 256, 0, stream>>>(Wh, whbf, (int)nW);
  k_f32_to_bf16<<<(int)(nW / 256), 256, 0, stream>>>(Wx, wxbf, (int)nW);
  k_f32_to_bf16<<<(int)(nSeq / 256), 256, 0, stream>>>(seq, seqbf, (int)nSeq);

  // zero c / hbf / h states: contiguous 786432 bytes = 196608 dwords
  k_zero_u32<<<196608 / 256, 256, 0, stream>>>((uint32_t*)cst, 196608);

  uint16_t* hbf0[2] = { hbfb,          hbfb + BH };
  uint16_t* hbf1[2] = { hbfb + 2 * BH, hbfb + 3 * BH };
  const size_t nWl = nW / 2;                  // per-layer weight slice

  for (int t = 0; t < T; ++t) {
    const int rd = t & 1, wr = rd ^ 1;
    // layer 0: x = sequence[:, t, :]
    k_lstm_layer_step<<<64, 256, 0, stream>>>(
        whbf, wxbf, hbf0[rd], seqbf + (size_t)t * I, (long)T * I,
        bias, cst, hst, hbf0[wr]);
    // layer 1: x = h of layer 0 (just written)
    k_lstm_layer_step<<<64, 256, 0, stream>>>(
        whbf + nWl, wxbf + nWl, hbf1[rd], hbf0[wr], (long)H,
        bias + 4 * H, cst + BH, hst + BH, hbf1[wr]);
  }

  // output = final h of last layer, fp32 (32 x 1024)
  hipMemcpyAsync(d_out, hst + BH, (size_t)BH * sizeof(float),
                 hipMemcpyDeviceToDevice, stream);
}
